// WindowAttention_7249904795781
// MI455X (gfx1250) — compile-verified, matrix operand order checked by reading, not run on hardware
//
#include <hip/hip_runtime.h>
#include <hip/hip_bf16.h>

typedef __attribute__((ext_vector_type(16))) __bf16 v16bf;
typedef __attribute__((ext_vector_type(8)))  float  v8f;

#define WMMA_BF16(a, b, c) \
  __builtin_amdgcn_wmma_f32_16x16x32_bf16(false, (a), false, (b), (short)0, (c), false, false)

__device__ __forceinline__ __bf16 f2bf(float f) { return (__bf16)f; }

// gfx1250 async global->LDS copy (ASYNCcnt tracked, no VGPR round-trip)
__device__ __forceinline__ void async_b128(unsigned lds_off, const void* gptr) {
  asm volatile("global_load_async_to_lds_b128 %0, %1, off"
               :: "v"(lds_off), "v"((unsigned long long)(size_t)gptr) : "memory");
}
__device__ __forceinline__ void wait_async_all() {
  asm volatile("s_wait_asynccnt 0x0" ::: "memory");
}
__device__ __forceinline__ void wait_async_4() {
  asm volatile("s_wait_asynccnt 0x4" ::: "memory");
}

// A-frag 16x32 from row-major [M x K] tile in LDS; also serves as B-frag when
// the B source is row-major [N x K] (i.e. W stored [out x in]).
__device__ __forceinline__ v16bf ld_frag_rowmajor(const __bf16* base, int ldm, int lane) {
  const int r  = lane & 15;
  const int kb = (lane >> 4) << 3;
  const __bf16* p = base + r * ldm + kb;
  v16bf f;
#pragma unroll
  for (int e = 0; e < 8; ++e) { f[e] = p[e]; f[e + 8] = p[e + 16]; }
  return f;
}

// B-frag 32x16 from row-major [K x N] tile in LDS (column n = lane&15)
__device__ __forceinline__ v16bf ld_frag_kn(const __bf16* base, int ldm, int lane) {
  const int n  = lane & 15;
  const int kb = (lane >> 4) << 3;
  const __bf16* p = base + kb * ldm + n;
  v16bf f;
#pragma unroll
  for (int e = 0; e < 8; ++e) { f[e] = p[e * ldm]; f[e + 8] = p[(e + 16) * ldm]; }
  return f;
}

// ---------------------------------------------------------------------------
// fp32 -> bf16 one-shot conversion (x, qkv_w, proj_w)
// ---------------------------------------------------------------------------
__global__ __launch_bounds__(256)
void f32_to_bf16_kernel(const float* __restrict__ in, __bf16* __restrict__ out, int n) {
  int i = blockIdx.x * 256 + threadIdx.x;
  const int stride = gridDim.x * 256;
  for (; i < n; i += stride) out[i] = f2bf(in[i]);
}

// ---------------------------------------------------------------------------
// Tiled GEMM:  Out[M x N] = A[M x K] @ B[N x K]^T + bias[N]   (A,B bf16)
// BM=BN=128, BK=32; 256 thr = 8 waves; wave tile 32x64; double-buffered
// async global->LDS staging overlapped with WMMA.
// ---------------------------------------------------------------------------
template <bool OUT_IS_BF16>
__global__ __launch_bounds__(256)
void gemm_bf16_kernel(const __bf16* __restrict__ A, const __bf16* __restrict__ B,
                      const float* __restrict__ bias, void* __restrict__ Out,
                      int M, int N, int K) {
  constexpr int BM = 128, BN = 128, BK = 32, LDT = BK + 8;  // +8 bf16 pads banks
  __shared__ __align__(16) __bf16 sA[2][BM * LDT];
  __shared__ __align__(16) __bf16 sB[2][BN * LDT];

  const int tid  = threadIdx.x;
  const int lane = tid & 31;
  const int wv   = tid >> 5;   // 0..7
  const int wm   = wv & 3;     // 4 wave rows  (32 rows each)
  const int wn   = wv >> 2;    // 2 wave cols  (64 cols each)
  const int bm   = blockIdx.y * BM;
  const int bn   = blockIdx.x * BN;

  // stage one 128x32 bf16 tile pair: 4 x b128 async copies per thread
  auto issue = [&](int k0, int buf) {
#pragma unroll
    for (int i = 0; i < 2; ++i) {
      const int e = tid + i * 256;          // 0..511
      const int r = e >> 2, c = e & 3;      // row 0..127, 16B chunk 0..3
      async_b128((unsigned)(size_t)&sA[buf][r * LDT + c * 8],
                 A + (size_t)(bm + r) * K + k0 + c * 8);
    }
#pragma unroll
    for (int i = 0; i < 2; ++i) {
      const int e = tid + i * 256;
      const int r = e >> 2, c = e & 3;
      async_b128((unsigned)(size_t)&sB[buf][r * LDT + c * 8],
                 B + (size_t)(bn + r) * K + k0 + c * 8);
    }
  };

  v8f zero = {};
  v8f acc[2][4];
#pragma unroll
  for (int i = 0; i < 2; ++i)
#pragma unroll
    for (int j = 0; j < 4; ++j) acc[i][j] = zero;

  const int nsteps = K / BK;  // 12
  issue(0, 0);
  for (int k = 0; k < nsteps; ++k) {
    const int buf = k & 1;
    if (k + 1 < nsteps) { issue((k + 1) * BK, buf ^ 1); wait_async_4(); }
    else                { wait_async_all(); }
    __syncthreads();  // all waves' tiles for step k resident

    v16bf afr[2], bfr[4];
#pragma unroll
    for (int i = 0; i < 2; ++i)
      afr[i] = ld_frag_rowmajor(&sA[buf][(wm * 32 + i * 16) * LDT], LDT, lane);
#pragma unroll
    for (int j = 0; j < 4; ++j)
      bfr[j] = ld_frag_rowmajor(&sB[buf][(wn * 64 + j * 16) * LDT], LDT, lane);
#pragma unroll
    for (int i = 0; i < 2; ++i)
#pragma unroll
      for (int j = 0; j < 4; ++j)
        acc[i][j] = WMMA_BF16(afr[i], bfr[j], acc[i][j]);
    __syncthreads();  // done reading buf before it is re-issued
  }

  // epilogue: D layout row = jj + 8*(lane>=16), col = lane&15
  const int rl = lane & 15;
  const int rh = (lane >> 4) << 3;
#pragma unroll
  for (int i = 0; i < 2; ++i) {
#pragma unroll
    for (int j = 0; j < 4; ++j) {
      const int col = bn + wn * 64 + j * 16 + rl;
      const float bv = bias[col];
#pragma unroll
      for (int jj = 0; jj < 8; ++jj) {
        const int row = bm + wm * 32 + i * 16 + jj + rh;
        const float v = acc[i][j][jj] + bv;
        if (OUT_IS_BF16) ((__bf16*)Out)[(size_t)row * N + col] = f2bf(v);
        else             ((float*)Out)[(size_t)row * N + col]  = v;
      }
    }
  }
}

// ---------------------------------------------------------------------------
// Window attention per (window b, head h). 128 threads = 4 waves.
// q/k/v [64x32] bf16 staged with async copies (rows 49..63 zeroed via ds);
// S = q@k^T; softmax(+rel-pos bias); O = P@v with deferred 1/rowsum.
// ---------------------------------------------------------------------------
__global__ __launch_bounds__(128)
void win_attn_kernel(const __bf16* __restrict__ qkv,       // [B*49, 1152]
                     const float* __restrict__ bias_table, // [169, 12]
                     const int*   __restrict__ rel_index,  // [49, 49]
                     __bf16* __restrict__ aout) {          // [B*49, 384]
  constexpr int NT = 64, HD = 32, NW = 49, NH = 12;
  __shared__ __align__(16) __bf16 sq[NT * HD], sk[NT * HD], sv[NT * HD];
  __shared__ float  sS[NT * NT];
  __shared__ __align__(16) __bf16 sP[NT * NT];
  __shared__ float  sInv[NT];

  const int tid  = threadIdx.x;
  const int lane = tid & 31;
  const int wv   = tid >> 5;          // 0..3, each owns 16 query rows
  const int b    = blockIdx.x / NH;
  const int h    = blockIdx.x % NH;
  const size_t rowbase = (size_t)b * NW * 1152;
  const int coff = h * HD;

  // zero padding rows 49..63 of q/k/v (disjoint from async destinations)
  for (int i = tid; i < 3 * 240; i += 128) {        // 15 rows * 32 elems / 2 per u32
    const int m = i / 240, rem = i - m * 240;
    __bf16* base = (m == 0) ? sq : (m == 1) ? sk : sv;
    ((unsigned*)(base + NW * HD))[rem] = 0u;
  }
  // async-stage rows < 49: 4 x 16B chunks per row per matrix
  for (int e = tid; e < 3 * 196; e += 128) {
    const int m = e / 196, rem = e - m * 196;
    const int r = rem >> 2, c = rem & 3;
    __bf16* base = (m == 0) ? sq : (m == 1) ? sk : sv;
    async_b128((unsigned)(size_t)(base + r * HD + c * 8),
               qkv + rowbase + (size_t)r * 1152 + m * 384 + coff + c * 8);
  }
  wait_async_all();
  __syncthreads();

  // S = q @ k^T : wave wv computes rows [16wv, 16wv+16) x all 64 cols
  {
    const v16bf aq = ld_frag_rowmajor(&sq[wv * 16 * HD], HD, lane);
    const int rl = lane & 15;
    const int rh = (lane >> 4) << 3;
#pragma unroll
    for (int t = 0; t < 4; ++t) {
      v16bf bk = ld_frag_rowmajor(&sk[t * 16 * HD], HD, lane);
      v8f c = {};
      c = WMMA_BF16(aq, bk, c);
#pragma unroll
      for (int jj = 0; jj < 8; ++jj)
        sS[(wv * 16 + jj + rh) * NT + t * 16 + rl] = c[jj];
    }
  }
  __syncthreads();

  // softmax per row (deferred normalization: store exp(), keep 1/sum)
  if (tid < NT) {
    const int n = tid;
    float inv = 0.f;
    if (n < NW) {
      const float scale = 0.17677669529663687f;  // 32^-0.5
      float mx = -3.0e38f;
      for (int m = 0; m < NT; ++m) {
        float s = sS[n * NT + m] * scale;
        s += (m < NW) ? bias_table[rel_index[n * NW + m] * NH + h] : -1.0e30f;
        sS[n * NT + m] = s;
        mx = fmaxf(mx, s);
      }
      float sum = 0.f;
      for (int m = 0; m < NT; ++m) {
        float e = __expf(sS[n * NT + m] - mx);
        sum += e;
        sP[n * NT + m] = f2bf(e);
      }
      inv = 1.0f / sum;
    } else {
      for (int m = 0; m < NT; ++m) sP[n * NT + m] = (__bf16)0.f;
    }
    sInv[n] = inv;
  }
  __syncthreads();

  // O = P @ V : K=64 (2 chunks of 32), 32 output cols (2 tiles of 16)
  v8f oa0 = {}, oa1 = {};
#pragma unroll
  for (int kc = 0; kc < 2; ++kc) {
    const v16bf ap = ld_frag_rowmajor(&sP[(wv * 16) * NT + kc * 32], NT, lane);
    v16bf bv0 = ld_frag_kn(&sv[(kc * 32) * HD + 0],  HD, lane);
    v16bf bv1 = ld_frag_kn(&sv[(kc * 32) * HD + 16], HD, lane);
    oa0 = WMMA_BF16(ap, bv0, oa0);
    oa1 = WMMA_BF16(ap, bv1, oa1);
  }

  const int rl = lane & 15;
  const int rh = (lane >> 4) << 3;
#pragma unroll
  for (int jj = 0; jj < 8; ++jj) {
    const int r = wv * 16 + jj + rh;
    if (r < NW) {
      const float sc = sInv[r];
      const size_t orow = ((size_t)b * NW + r) * 384 + coff;
      aout[orow + rl]      = f2bf(oa0[jj] * sc);
      aout[orow + 16 + rl] = f2bf(oa1[jj] * sc);
    }
  }
}

// ---------------------------------------------------------------------------
extern "C" void kernel_launch(void* const* d_in, const int* in_sizes, int n_in,
                              void* d_out, int out_size, void* d_ws, size_t ws_size,
                              hipStream_t stream) {
  (void)in_sizes; (void)n_in; (void)out_size; (void)ws_size;
  const float* x       = (const float*)d_in[0];
  const float* qkv_w   = (const float*)d_in[1];  // [1152, 384]
  const float* qkv_b   = (const float*)d_in[2];  // [1152]
  const float* proj_w  = (const float*)d_in[3];  // [384, 384]
  const float* proj_b  = (const float*)d_in[4];  // [384]
  const float* bias_tb = (const float*)d_in[5];  // [169, 12]
  const int*   rel_idx = (const int*)d_in[6];    // [49, 49]

  const int Bw = 4096, N = 49, H = 12, C = 384;
  const int M = Bw * N;  // 200704 (divisible by 128)

  // workspace layout (all offsets 256B-aligned)
  char* ws = (char*)d_ws;
  __bf16* qkvbuf = (__bf16*)(ws);                                 // [M x 1152]
  __bf16* aobuf  = (__bf16*)(ws + (size_t)M * 1152 * 2);          // [M x 384]
  __bf16* xbf    = (__bf16*)(ws + (size_t)M * 1536 * 2);          // [M x 384]
  __bf16* qwbf   = (__bf16*)(ws + (size_t)M * 1920 * 2);          // [1152 x 384]
  __bf16* pwbf   = (__bf16*)(ws + (size_t)M * 1920 * 2 + 1152 * 384 * 2);

  // 0) one-shot fp32 -> bf16 conversions
  {
    const int nx = M * C;  // 77,070,336
    f32_to_bf16_kernel<<<(nx + 2047) / 2048, 256, 0, stream>>>(x, xbf, nx);
    f32_to_bf16_kernel<<<(1152 * 384 + 2047) / 2048, 256, 0, stream>>>(qkv_w, qwbf, 1152 * 384);
    f32_to_bf16_kernel<<<(384 * 384 + 2047) / 2048, 256, 0, stream>>>(proj_w, pwbf, 384 * 384);
  }
  // 1) qkv = x @ qkv_w^T + qkv_b   (bf16 out)
  {
    dim3 grid(1152 / 128, M / 128);
    gemm_bf16_kernel<true><<<grid, 256, 0, stream>>>(xbf, qwbf, qkv_b, (void*)qkvbuf, M, 1152, 384);
  }
  // 2) window attention per (window, head)
  win_attn_kernel<<<Bw * H, 128, 0, stream>>>(qkvbuf, bias_tb, rel_idx, aobuf);
  // 3) out = ao @ proj_w^T + proj_b   (fp32 out)
  {
    dim3 grid(384 / 128, M / 128);
    gemm_bf16_kernel<false><<<grid, 256, 0, stream>>>(aobuf, pwbf, proj_b, d_out, M, 384, 384);
  }
}
